// RAFT_32512902431052
// MI455X (gfx1250) — compile-verified, matrix operand order checked by reading, not run on hardware
//
#include <hip/hip_runtime.h>
#include <cmath>

// ---------------------------------------------------------------------------
// CDNA5 (gfx1250) RAFT-small forward.
//   * bf16 WMMA (v_wmma_f32_16x16x32_bf16) for correlation GEMMs and all
//     convolutions (implicit GEMM), fp32 accumulate, 2x N register blocking.
//   * All LDS fragment traffic is 16-byte (ds_load_b128/ds_store_b128):
//     A and B tiles are staged K-contiguous per row with 40-element padded
//     rows (20-bank stride -> conflict-free across the 16 lane rows).
//   * Plain VALU kernels for pooling / bilinear lookup / GRU pointwise /
//     align-corners upsampling (gather- or bandwidth-bound).
// ---------------------------------------------------------------------------

typedef __attribute__((ext_vector_type(16))) __bf16 bf16x16;
typedef __attribute__((ext_vector_type(8)))  __bf16 bf16x8;
typedef __attribute__((ext_vector_type(8)))  float  floatx8;

#define ACT_NONE    0
#define ACT_RELU    1
#define ACT_SIGMOID 2
#define ACT_TANH    3

#define LDSPITCH 40   // 32 payload + 8 pad bf16 -> 80B row stride (16B aligned)

__device__ inline bf16x16 frag_load(const __bf16* row, int half) {
  // 16 K-elements per lane = two contiguous 16B chunks per ISA 16-bit layout
  bf16x8 lo = *(const bf16x8*)(row + half * 8);
  bf16x8 hi = *(const bf16x8*)(row + 16 + half * 8);
  return __builtin_shufflevector(lo, hi, 0, 1, 2, 3, 4, 5, 6, 7,
                                 8, 9, 10, 11, 12, 13, 14, 15);
}

// ---------------- conversion kernels ----------------

__global__ void k_cvt_bf16(const float* __restrict__ src, __bf16* __restrict__ dst,
                           int n, float scale) {
  int i = blockIdx.x * blockDim.x + threadIdx.x;
  if (i < n) dst[i] = (__bf16)(src[i] * scale);
}

// src [C, HW] (NCHW flattened) -> dst [HW, C]  (K-contiguous rows)
__global__ void k_cvt_transpose_bf16(const float* __restrict__ src, __bf16* __restrict__ dst,
                                     int C, int HW, float scale) {
  int i = blockIdx.x * blockDim.x + threadIdx.x;
  if (i < C * HW) {
    int c = i / HW, p = i % HW;
    dst[(size_t)p * C + c] = (__bf16)(src[i] * scale);
  }
}

// ---------------- correlation GEMM: D[M,N] = A[M,128] x Bt[N,128]^T --------
// A and Bt share the [*,128] bf16 K-contiguous layout. 256 threads = 8 waves,
// block tile 64x64; waves 4x2, each wave computes two adjacent 16x16 tiles
// (A fragment reused across both v_wmma). M,N multiples of 64 at all scales.

__global__ void __launch_bounds__(256)
k_corr_wmma(const __bf16* __restrict__ A, const __bf16* __restrict__ Bt,
            float* __restrict__ D, int M, int N) {
  __shared__ __align__(16) __bf16 As[64][LDSPITCH];
  __shared__ __align__(16) __bf16 BsT[64][LDSPITCH];
  const int K = 128;
  const int tid  = threadIdx.x;
  const int wave = tid >> 5;
  const int lane = tid & 31;
  const int wm = wave & 3;         // 4 M sub-tiles
  const int wn = wave >> 2;        // 2 N sub-tiles of 32
  const int m0 = blockIdx.y * 64;
  const int n0 = blockIdx.x * 64;
  const int ln   = lane & 15;
  const int half = lane >> 4;
  const int sr = tid >> 2;         // staging row 0..63 (one 16B chunk each)
  const int sc = (tid & 3) * 8;    // staging chunk offset in K

  floatx8 acc0 = {}, acc1 = {};
  for (int k0 = 0; k0 < K; k0 += 32) {
    *(bf16x8*)&As[sr][sc]  = *(const bf16x8*)(A  + (size_t)(m0 + sr) * K + (k0 + sc));
    *(bf16x8*)&BsT[sr][sc] = *(const bf16x8*)(Bt + (size_t)(n0 + sr) * K + (k0 + sc));
    if (k0 + 32 < K) {  // gfx1250 global_prefetch_b8 on the next K slice
      __builtin_prefetch(A  + (size_t)(m0 + sr) * K + (k0 + 32 + sc), 0, 1);
      __builtin_prefetch(Bt + (size_t)(n0 + sr) * K + (k0 + 32 + sc), 0, 1);
    }
    __syncthreads();

    bf16x16 afrag = frag_load(&As[wm * 16 + ln][0], half);
    bf16x16 b0    = frag_load(&BsT[wn * 32 + ln][0], half);
    bf16x16 b1    = frag_load(&BsT[wn * 32 + 16 + ln][0], half);
    acc0 = __builtin_amdgcn_wmma_f32_16x16x32_bf16(false, afrag, false, b0,
                                                   (short)0, acc0, false, false);
    acc1 = __builtin_amdgcn_wmma_f32_16x16x32_bf16(false, afrag, false, b1,
                                                   (short)0, acc1, false, false);
    __syncthreads();
  }
#pragma unroll
  for (int v = 0; v < 8; ++v) {
    int m = m0 + wm * 16 + v + ((lane >= 16) ? 8 : 0);
    int n = n0 + wn * 32 + ln;
    D[(size_t)m * N + n]      = acc0[v];
    D[(size_t)m * N + n + 16] = acc1[v];
  }
}

// ---------------- implicit-GEMM convolution with WMMA ----------------------
// Y[co*HW + p] = act( sum_k W[co,k] * im2col(k,p) + bias[co] )
// Block tile 32x128, 8 waves 2x4, each wave two adjacent 16x16 tiles.
// im2col chunks are built in registers (8 gathers) -> one ds_store_b128.

__global__ void __launch_bounds__(256)
k_conv_wmma(const __bf16* __restrict__ Wm, const float* __restrict__ bias,
            const float* __restrict__ X, float* __restrict__ Y,
            int Co, int Ci, int H, int Wd, int kh, int kw, int pad, int act) {
  __shared__ __align__(16) __bf16 As[32][LDSPITCH];
  __shared__ __align__(16) __bf16 BsT[128][LDSPITCH];
  const int HW  = H * Wd;
  const int khw = kh * kw;
  const int K   = Ci * khw;
  const int tid  = threadIdx.x;
  const int wave = tid >> 5;
  const int lane = tid & 31;
  const int wm = wave & 1;         // 2 M sub-tiles
  const int wn = wave >> 1;        // 4 N sub-tiles of 32
  const int m0 = blockIdx.y * 32;
  const int n0 = blockIdx.x * 128;
  const int ln   = lane & 15;
  const int half = lane >> 4;

  floatx8 acc0 = {}, acc1 = {};
  for (int k0 = 0; k0 < K; k0 += 32) {
    if (tid < 128) {               // weight tile: 32 rows x 4 chunks
      int r = tid >> 2, c4 = (tid & 3) * 8;
      int gm = m0 + r;
      bf16x8 v;
#pragma unroll
      for (int i = 0; i < 8; ++i) {
        int gk = k0 + c4 + i;
        v[i] = (gm < Co && gk < K) ? Wm[(size_t)gm * K + gk] : (__bf16)0.0f;
      }
      *(bf16x8*)&As[r][c4] = v;
    }
    for (int t = tid; t < 512; t += 256) {   // im2col tile: 128 n x 4 chunks
      int n = t >> 2, c4 = (t & 3) * 8;
      int gp = n0 + n;
      int oy = gp / Wd, ox = gp % Wd;
      bf16x8 v;
#pragma unroll
      for (int i = 0; i < 8; ++i) {
        int gk = k0 + c4 + i;
        float val = 0.0f;
        if (gk < K && gp < HW) {
          int ci = gk / khw, rem = gk % khw;
          int ky = rem / kw, kx = rem % kw;
          int iy = oy + ky - pad, ix = ox + kx - pad;
          if (iy >= 0 && iy < H && ix >= 0 && ix < Wd)
            val = X[(size_t)ci * HW + iy * Wd + ix];
        }
        v[i] = (__bf16)val;
      }
      *(bf16x8*)&BsT[n][c4] = v;
    }
    __syncthreads();

    bf16x16 afrag = frag_load(&As[wm * 16 + ln][0], half);
    bf16x16 b0    = frag_load(&BsT[wn * 32 + ln][0], half);
    bf16x16 b1    = frag_load(&BsT[wn * 32 + 16 + ln][0], half);
    acc0 = __builtin_amdgcn_wmma_f32_16x16x32_bf16(false, afrag, false, b0,
                                                   (short)0, acc0, false, false);
    acc1 = __builtin_amdgcn_wmma_f32_16x16x32_bf16(false, afrag, false, b1,
                                                   (short)0, acc1, false, false);
    __syncthreads();
  }
#pragma unroll
  for (int v = 0; v < 8; ++v) {
    int m = m0 + wm * 16 + v + ((lane >= 16) ? 8 : 0);
    int n = n0 + wn * 32 + ln;
    if (m < Co) {
      float b = bias[m];
#pragma unroll
      for (int h2 = 0; h2 < 2; ++h2) {
        int nn = n + h2 * 16;
        if (nn < HW) {
          float val = (h2 ? acc1[v] : acc0[v]) + b;
          if      (act == ACT_RELU)    val = fmaxf(val, 0.0f);
          else if (act == ACT_SIGMOID) val = 1.0f / (1.0f + __expf(-val));
          else if (act == ACT_TANH)    val = tanhf(val);
          Y[(size_t)m * HW + nn] = val;
        }
      }
    }
  }
}

// ---------------- pyramid / lookup ----------------

__global__ void k_avgpool(const float* __restrict__ in, float* __restrict__ out,
                          int Nimg, int H, int W) {
  int Ho = H >> 1, Wo = W >> 1;
  size_t total = (size_t)Nimg * Ho * Wo;
  size_t i = (size_t)blockIdx.x * blockDim.x + threadIdx.x;
  if (i >= total) return;
  int x = (int)(i % Wo);
  size_t t = i / Wo;
  int y = (int)(t % Ho);
  int img = (int)(t / Ho);
  const float* p = in + (size_t)img * H * W;
  out[i] = 0.25f * (p[(2 * y) * W + 2 * x]     + p[(2 * y) * W + 2 * x + 1] +
                    p[(2 * y + 1) * W + 2 * x] + p[(2 * y + 1) * W + 2 * x + 1]);
}

__device__ inline float bilin_zero(const float* __restrict__ img, int H, int W,
                                   float x, float y) {
  float x0f = floorf(x), y0f = floorf(y);
  int x0 = (int)x0f, y0 = (int)y0f;
  float wx = x - x0f, wy = y - y0f;
  float acc = 0.0f;
#pragma unroll
  for (int dy = 0; dy < 2; ++dy)
#pragma unroll
    for (int dx = 0; dx < 2; ++dx) {
      int xi = x0 + dx, yi = y0 + dy;
      if (xi >= 0 && xi < W && yi >= 0 && yi < H) {
        float w = (dx ? wx : 1.0f - wx) * (dy ? wy : 1.0f - wy);
        acc += w * img[yi * W + xi];
      }
    }
  return acc;
}

__global__ void k_lookup(const float* __restrict__ c1,
                         const float* __restrict__ l0, const float* __restrict__ l1,
                         const float* __restrict__ l2, const float* __restrict__ l3,
                         float* __restrict__ out, int H, int W) {
  int HW = H * W;
  int p = blockIdx.x * blockDim.x + threadIdx.x;
  if (p >= HW) return;
  float cx = c1[p], cy = c1[HW + p];
  const float* lv[4] = {l0, l1, l2, l3};
  for (int lev = 0; lev < 4; ++lev) {
    int Hl = H >> lev, Wl = W >> lev;
    const float* img = lv[lev] + (size_t)p * Hl * Wl;
    float inv = 1.0f / (float)(1 << lev);
    float bx = cx * inv, by = cy * inv;
    for (int dyi = 0; dyi < 7; ++dyi)
      for (int dxi = 0; dxi < 7; ++dxi) {
        // reference adds stack(meshgrid(dy,dx)) to (x,y)
        float sx = bx + (float)(dyi - 3);
        float sy = by + (float)(dxi - 3);
        out[(size_t)(lev * 49 + dyi * 7 + dxi) * HW + p] = bilin_zero(img, Hl, Wl, sx, sy);
      }
  }
}

// ---------------- pointwise / state kernels ----------------

__global__ void k_tanh(const float* __restrict__ s, float* __restrict__ d, int n) {
  int i = blockIdx.x * blockDim.x + threadIdx.x;
  if (i < n) d[i] = tanhf(s[i]);
}
__global__ void k_relu(const float* __restrict__ s, float* __restrict__ d, int n) {
  int i = blockIdx.x * blockDim.x + threadIdx.x;
  if (i < n) d[i] = fmaxf(s[i], 0.0f);
}
__global__ void k_copy(const float* __restrict__ s, float* __restrict__ d, int n) {
  int i = blockIdx.x * blockDim.x + threadIdx.x;
  if (i < n) d[i] = s[i];
}
__global__ void k_mul(const float* __restrict__ a, const float* __restrict__ b,
                      float* __restrict__ d, int n) {
  int i = blockIdx.x * blockDim.x + threadIdx.x;
  if (i < n) d[i] = a[i] * b[i];
}
__global__ void k_gru(const float* __restrict__ z, const float* __restrict__ q,
                      float* __restrict__ net, int n) {
  int i = blockIdx.x * blockDim.x + threadIdx.x;
  if (i < n) net[i] = (1.0f - z[i]) * net[i] + z[i] * q[i];
}
__global__ void k_add(float* __restrict__ a, const float* __restrict__ b, int n) {
  int i = blockIdx.x * blockDim.x + threadIdx.x;
  if (i < n) a[i] += b[i];
}
__global__ void k_initcoords(float* __restrict__ c, int H, int W) {
  int HW = H * W;
  int p = blockIdx.x * blockDim.x + threadIdx.x;
  if (p >= HW) return;
  c[p]      = (float)(p % W);
  c[HW + p] = (float)(p / W);
}
__global__ void k_flow(const float* __restrict__ c1, float* __restrict__ flow, int H, int W) {
  int HW = H * W;
  int p = blockIdx.x * blockDim.x + threadIdx.x;
  if (p >= HW) return;
  flow[p]      = c1[p]      - (float)(p % W);
  flow[HW + p] = c1[HW + p] - (float)(p / W);
}

// bilinear resize (align_corners=True, interp_mat-compatible clamp) * mult
__global__ void k_upsample(const float* __restrict__ in, int C, int Hi, int Wi,
                           float* __restrict__ out, int Ho, int Wo, float mult) {
  int total = C * Ho * Wo;
  int i = blockIdx.x * blockDim.x + threadIdx.x;
  if (i >= total) return;
  int x = i % Wo;
  int t = i / Wo;
  int y = t % Ho;
  int c = t / Ho;
  float py = (Ho > 1) ? (float)y * (float)(Hi - 1) / (float)(Ho - 1) : 0.0f;
  float px = (Wo > 1) ? (float)x * (float)(Wi - 1) / (float)(Wo - 1) : 0.0f;
  int y0 = (int)floorf(py); if (y0 > Hi - 2) y0 = Hi - 2; if (y0 < 0) y0 = 0;
  int x0 = (int)floorf(px); if (x0 > Wi - 2) x0 = Wi - 2; if (x0 < 0) x0 = 0;
  float wy = py - (float)y0, wx = px - (float)x0;
  const float* img = in + (size_t)c * Hi * Wi;
  float v = (1.0f - wy) * ((1.0f - wx) * img[y0 * Wi + x0] + wx * img[y0 * Wi + x0 + 1]) +
            wy          * ((1.0f - wx) * img[(y0 + 1) * Wi + x0] + wx * img[(y0 + 1) * Wi + x0 + 1]);
  out[i] = mult * v;
}

// ---------------------------------------------------------------------------

extern "C" void kernel_launch(void* const* d_in, const int* in_sizes, int n_in,
                              void* d_out, int out_size, void* d_ws, size_t ws_size,
                              hipStream_t stream) {
  (void)in_sizes; (void)n_in; (void)out_size; (void)ws_size;

  // si: 0 -> s=2 (96x96), 1 -> s=4 (48x48), 2 -> s=8 (24x24)
  const float* fmap1[3] = {(const float*)d_in[0], (const float*)d_in[2], (const float*)d_in[4]};
  const float* fmap2[3] = {(const float*)d_in[1], (const float*)d_in[3], (const float*)d_in[5]};
  const float* net_in[3] = {(const float*)d_in[6], (const float*)d_in[8], (const float*)d_in[10]};
  const float* inp_in[3] = {(const float*)d_in[7], (const float*)d_in[9], (const float*)d_in[11]};

  // conv weights: {convc1, convf1, convf2, convm, convz, convr, convq, fh1, fh2}
  const int wco[9] = {96, 64, 32, 80, 96, 96, 96, 128, 2};
  const int wk [9] = {196, 2 * 49, 64 * 9, 128 * 9, 242 * 9, 242 * 9, 242 * 9, 96 * 9, 128 * 9};
  const float* wsrc[9]; const float* bsrc[9];
  for (int i = 0; i < 9; ++i) {
    wsrc[i] = (const float*)d_in[12 + 2 * i];
    bsrc[i] = (const float*)d_in[13 + 2 * i];
  }

  const int Hs[3] = {96, 48, 24};
  char* ws = (char*)d_ws;
  size_t off = 0;
  auto alloc = [&](size_t bytes) -> size_t {
    size_t o = off; off += (bytes + 255) & ~(size_t)255; return o;
  };
  auto F  = [&](size_t o) -> float*  { return (float*)(ws + o); };
  auto BF = [&](size_t o) -> __bf16* { return (__bf16*)(ws + o); };

  size_t wbf_off[9];
  for (int i = 0; i < 9; ++i) wbf_off[i] = alloc((size_t)wco[i] * wk[i] * sizeof(__bf16));

  size_t pyr_off[3][4], c1_off[3], net_off[3], inp_off[3];
  for (int si = 0; si < 3; ++si) {
    int H = Hs[si]; size_t HW = (size_t)H * H;
    for (int l = 0; l < 4; ++l)
      pyr_off[si][l] = alloc(HW * (size_t)(H >> l) * (H >> l) * sizeof(float));
    c1_off[si]  = alloc(2  * HW * sizeof(float));
    net_off[si] = alloc(96 * HW * sizeof(float));
    inp_off[si] = alloc(64 * HW * sizeof(float));
  }
  const size_t HW0 = 96 * 96;  // transient buffers sized for the largest scale
  size_t f1T_off   = alloc(HW0 * 128 * sizeof(__bf16));
  size_t f2T_off   = alloc(HW0 * 128 * sizeof(__bf16));
  size_t corrf_off = alloc(196 * HW0 * sizeof(float));
  size_t flow_off  = alloc(2   * HW0 * sizeof(float));
  size_t flo1_off  = alloc(64  * HW0 * sizeof(float));
  size_t mcat_off  = alloc(128 * HW0 * sizeof(float));   // [cor(96) | flo(32)]
  size_t cat_off   = alloc(242 * HW0 * sizeof(float));   // [head(96)|inp(64)|mot(80)|flow(2)]
  size_t z_off     = alloc(96  * HW0 * sizeof(float));
  size_t r_off     = alloc(96  * HW0 * sizeof(float));
  size_t q_off     = alloc(96  * HW0 * sizeof(float));
  size_t fh_off    = alloc(128 * HW0 * sizeof(float));
  size_t df_off    = alloc(2   * HW0 * sizeof(float));

  dim3 blk(256);
  auto g1 = [](size_t n) { return dim3((unsigned)((n + 255) / 256)); };

  // ---- weights -> bf16 (once per call; deterministic) ----
  for (int i = 0; i < 9; ++i) {
    size_t n = (size_t)wco[i] * wk[i];
    k_cvt_bf16<<<g1(n), blk, 0, stream>>>(wsrc[i], BF(wbf_off[i]), (int)n, 1.0f);
  }

  // ---- per-scale setup: correlation pyramid + state init ----
  const float cscale = 1.0f / sqrtf(128.0f);
  for (int si = 0; si < 3; ++si) {
    int H = Hs[si]; int HW = H * H;
    k_cvt_transpose_bf16<<<g1((size_t)128 * HW), blk, 0, stream>>>(
        fmap1[si], BF(f1T_off), 128, HW, cscale);
    k_cvt_transpose_bf16<<<g1((size_t)128 * HW), blk, 0, stream>>>(
        fmap2[si], BF(f2T_off), 128, HW, 1.0f);
    dim3 cg(HW / 64, HW / 64);   // HW in {9216,2304,576}: exact tiling
    k_corr_wmma<<<cg, blk, 0, stream>>>(BF(f1T_off), BF(f2T_off),
                                        F(pyr_off[si][0]), HW, HW);
    for (int l = 0; l < 3; ++l) {
      int Hl = H >> l;
      size_t tot = (size_t)HW * (Hl / 2) * (Hl / 2);
      k_avgpool<<<g1(tot), blk, 0, stream>>>(F(pyr_off[si][l]), F(pyr_off[si][l + 1]),
                                             HW, Hl, Hl);
    }
    k_initcoords<<<g1((size_t)HW), blk, 0, stream>>>(F(c1_off[si]), H, H);
    k_tanh<<<g1((size_t)96 * HW), blk, 0, stream>>>(net_in[si], F(net_off[si]), 96 * HW);
    k_relu<<<g1((size_t)64 * HW), blk, 0, stream>>>(inp_in[si], F(inp_off[si]), 64 * HW);
  }

  // ---- coarse-to-fine update iterations ----
  float* out = (float*)d_out;
  const int seq[6] = {8, 8, 4, 4, 2, 2};
  for (int it = 0; it < 6; ++it) {
    int s  = seq[it];
    int si = (s == 2) ? 0 : ((s == 4) ? 1 : 2);
    int H = Hs[si]; int HW = H * H;
    float* c1   = F(c1_off[si]);
    float* netb = F(net_off[si]);

    k_lookup<<<g1((size_t)HW), blk, 0, stream>>>(
        c1, F(pyr_off[si][0]), F(pyr_off[si][1]), F(pyr_off[si][2]), F(pyr_off[si][3]),
        F(corrf_off), H, H);
    k_flow<<<g1((size_t)HW), blk, 0, stream>>>(c1, F(flow_off), H, H);

    auto conv = [&](int wi, const float* X, float* Y, int act, int kh, int pad) {
      int Co = wco[wi]; int Ci = wk[wi] / (kh * kh);
      dim3 g((HW + 127) / 128, (Co + 31) / 32);
      k_conv_wmma<<<g, blk, 0, stream>>>(BF(wbf_off[wi]), bsrc[wi], X, Y,
                                         Co, Ci, H, H, kh, kh, pad, act);
    };

    conv(0, F(corrf_off), F(mcat_off),                    ACT_RELU, 1, 0); // cor -> mcat[0:96]
    conv(1, F(flow_off),  F(flo1_off),                    ACT_RELU, 7, 3); // convf1
    conv(2, F(flo1_off),  F(mcat_off) + (size_t)96 * HW,  ACT_RELU, 3, 1); // flo -> mcat[96:128]
    conv(3, F(mcat_off),  F(cat_off) + (size_t)160 * HW,  ACT_RELU, 3, 1); // mot -> cat[160:240]
    k_copy<<<g1((size_t)64 * HW), blk, 0, stream>>>(F(inp_off[si]),
        F(cat_off) + (size_t)96 * HW, 64 * HW);                            // inp -> cat[96:160]
    k_copy<<<g1((size_t)2 * HW), blk, 0, stream>>>(F(flow_off),
        F(cat_off) + (size_t)240 * HW, 2 * HW);                            // flow -> cat[240:242]
    k_copy<<<g1((size_t)96 * HW), blk, 0, stream>>>(netb, F(cat_off), 96 * HW); // net head
    conv(4, F(cat_off), F(z_off), ACT_SIGMOID, 3, 1);
    conv(5, F(cat_off), F(r_off), ACT_SIGMOID, 3, 1);
    k_mul<<<g1((size_t)96 * HW), blk, 0, stream>>>(F(r_off), netb, F(cat_off), 96 * HW);
    conv(6, F(cat_off), F(q_off), ACT_TANH, 3, 1);
    k_gru<<<g1((size_t)96 * HW), blk, 0, stream>>>(F(z_off), F(q_off), netb, 96 * HW);
    conv(7, netb,      F(fh_off), ACT_RELU, 3, 1);
    conv(8, F(fh_off), F(df_off), ACT_NONE, 3, 1);

    k_add<<<g1((size_t)2 * HW), blk, 0, stream>>>(c1, F(df_off), 2 * HW);
    k_flow<<<g1((size_t)HW), blk, 0, stream>>>(c1, F(flow_off), H, H);
    k_upsample<<<g1((size_t)2 * 192 * 192), blk, 0, stream>>>(
        F(flow_off), 2, H, H, out + (size_t)it * 2 * 192 * 192, 192, 192, (float)s);
    if (s > 2) {
      int sj = si - 1;  // next finer scale
      k_upsample<<<g1((size_t)2 * (2 * H) * (2 * H)), blk, 0, stream>>>(
          c1, 2, H, H, F(c1_off[sj]), 2 * H, 2 * H, 2.0f);
    }
  }
}
